// Seq2Seq_Attn_72653666779205
// MI455X (gfx1250) — compile-verified, matrix operand order checked by reading, not run on hardware
//
#include <hip/hip_runtime.h>
#include <hip/hip_bf16.h>

// Seq2Seq+Attention for gfx1250 (MI455X). All GEMMs via v_wmma_f32_16x16x32_bf16
// (f32 accumulate). Weights pre-converted to bf16 (L2-resident); activations are
// f32 and converted to bf16 with packed ops while staging into LDS.

typedef __attribute__((ext_vector_type(16))) __bf16          v16bf;
typedef __attribute__((ext_vector_type(8)))  __bf16          v8bf;
typedef __attribute__((ext_vector_type(8)))  float           v8f;
typedef __attribute__((ext_vector_type(4)))  int             v4i;

#if defined(__has_builtin)
#if __has_builtin(__builtin_amdgcn_global_load_async_to_lds_b128)
#define USE_ASYNC_LDS 1
#endif
#endif

__device__ __forceinline__ void wait_async0() {
#if defined(__has_builtin) && __has_builtin(__builtin_amdgcn_s_wait_asynccnt)
  __builtin_amdgcn_s_wait_asynccnt(0);
#else
  asm volatile("s_wait_asynccnt 0x0" ::: "memory");
#endif
}

// pack two f32 into one dword of two bf16 ({b in [31:16], a in [15:0]})
__device__ __forceinline__ unsigned int pack2_bf16(float a, float b) {
#if defined(__has_builtin) && __has_builtin(__builtin_amdgcn_cvt_pk_bf16_f32)
  auto p = __builtin_amdgcn_cvt_pk_bf16_f32(a, b);   // v_cvt_pk_bf16_f32 (RNE)
  return __builtin_bit_cast(unsigned int, p);
#else
  unsigned int ua = __float_as_uint(a) + 0x8000u;    // round-to-nearest
  unsigned int ub = __float_as_uint(b) + 0x8000u;
  return __builtin_amdgcn_perm(ub, ua, 0x07060302u); // {ub[31:16], ua[31:16]}
#endif
}

__device__ __forceinline__ unsigned short f32_to_bf16_bits(float f) {
  unsigned int u = __float_as_uint(f);
  u += 0x7FFFu + ((u >> 16) & 1u);                   // RNE
  return (unsigned short)(u >> 16);
}

// ---------------------------------------------------------------------------
// WMMA GEMM phase: acc += A(MxKp) * W^T. W is bf16 (N rows, Kp cols, Kp%32==0,
// zero-padded). A is f32, lda%32==0, pad columns zero.
// Block tile 128x128, 256 threads = 8 wave32s in 4x2; each wave owns 32x64
// (2x4 accumulators -> 8 WMMA per K-chunk). LDS row stride 56 (112B: 16B-aligned
// fragment reads, conflict-free across the 16 rows a lane group touches).
// W tile staged with GLOBAL_LOAD_ASYNC_TO_LDS_B128 when available (ASYNCcnt).
// ---------------------------------------------------------------------------
__device__ __forceinline__ void gemm_phase(
    v8f acc[2][4],
    const float* __restrict__ A, int lda, int M,
    const unsigned short* __restrict__ W, int Kp, int N,
    int rowBase, int colBase,
    unsigned short* As, unsigned short* Bs)
{
  const int tid  = threadIdx.x;
  const int lane = tid & 31;
  const int wave = tid >> 5;
  const int wm   = wave >> 1;            // 0..3 : 32-row strip
  const int wn   = wave & 1;             // 0..1 : 64-col strip
  const int lm   = lane & 15;
  const int kh   = (lane & 16) ? 8 : 0;  // ISA 16-bit 16x32 K-half select

  for (int k0 = 0; k0 < Kp; k0 += 32) {
    { // stage A tile 128x32: f32 -> bf16 (packed), 16 elements/thread
      const int r  = tid >> 1;
      const int cb = (tid & 1) << 4;
      const int gm = rowBase + r;
      float v[16];
      if (gm < M) {
        const float* ap = A + (size_t)gm * lda + k0 + cb;
        #pragma unroll
        for (int q = 0; q < 4; ++q) {
          float4 f = *(const float4*)(ap + 4 * q);
          v[4*q+0] = f.x; v[4*q+1] = f.y; v[4*q+2] = f.z; v[4*q+3] = f.w;
        }
      } else {
        #pragma unroll
        for (int q = 0; q < 16; ++q) v[q] = 0.0f;
      }
      unsigned int w[8];
      #pragma unroll
      for (int q = 0; q < 8; ++q) w[q] = pack2_bf16(v[2*q], v[2*q+1]);
      uint4 lo = {w[0], w[1], w[2], w[3]};
      uint4 hi = {w[4], w[5], w[6], w[7]};
      *(uint4*)&As[r * 56 + cb]     = lo;
      *(uint4*)&As[r * 56 + cb + 8] = hi;
    }
    { // stage W tile 128x32 (bf16 direct), 16 elements/thread
      const int r  = tid >> 1;
      const int cb = (tid & 1) << 4;
      const int gn = colBase + r;
      if (gn < N) {
        const unsigned short* gp = W + (size_t)gn * Kp + k0 + cb;
#ifdef USE_ASYNC_LDS
        __builtin_amdgcn_global_load_async_to_lds_b128(
            (v4i*)gp, (v4i*)&Bs[r * 56 + cb], 0, 0);
        __builtin_amdgcn_global_load_async_to_lds_b128(
            (v4i*)(gp + 8), (v4i*)&Bs[r * 56 + cb + 8], 0, 0);
#else
        uint4 lo = *(const uint4*)gp;
        uint4 hi = *(const uint4*)(gp + 8);
        *(uint4*)&Bs[r * 56 + cb]     = lo;
        *(uint4*)&Bs[r * 56 + cb + 8] = hi;
#endif
      } else {
        uint4 zz = {0, 0, 0, 0};
        *(uint4*)&Bs[r * 56 + cb]     = zz;
        *(uint4*)&Bs[r * 56 + cb + 8] = zz;
      }
    }
#ifdef USE_ASYNC_LDS
    wait_async0();
#endif
    __syncthreads();

    v16bf af[2], bf2[4];
    #pragma unroll
    for (int i = 0; i < 2; ++i) {
      const int row = wm * 32 + i * 16 + lm;
      v8bf p0 = *(const v8bf*)&As[row * 56 + kh];
      v8bf p1 = *(const v8bf*)&As[row * 56 + kh + 16];
      af[i] = __builtin_shufflevector(p0, p1, 0,1,2,3,4,5,6,7,8,9,10,11,12,13,14,15);
    }
    #pragma unroll
    for (int j = 0; j < 4; ++j) {
      const int row = wn * 64 + j * 16 + lm;
      v8bf p0 = *(const v8bf*)&Bs[row * 56 + kh];
      v8bf p1 = *(const v8bf*)&Bs[row * 56 + kh + 16];
      bf2[j] = __builtin_shufflevector(p0, p1, 0,1,2,3,4,5,6,7,8,9,10,11,12,13,14,15);
    }
    #pragma unroll
    for (int i = 0; i < 2; ++i)
      #pragma unroll
      for (int j = 0; j < 4; ++j)
        acc[i][j] = __builtin_amdgcn_wmma_f32_16x16x32_bf16(
            false, af[i], false, bf2[j], (short)0, acc[i][j], false, false);
    __syncthreads();
  }
}

__global__ __launch_bounds__(256) void gemm_wmma_kernel(
    float* __restrict__ C, int ldc, int M, int N,
    const float* __restrict__ A1, int lda1, int K1, const unsigned short* __restrict__ W1,
    const float* __restrict__ A2, int lda2, int K2, const unsigned short* __restrict__ W2,
    const float* __restrict__ b1, const float* __restrict__ b2, int relu)
{
  __shared__ unsigned short As[128 * 56];
  __shared__ unsigned short Bs[128 * 56];
  const int rowBase = blockIdx.y * 128;
  const int colBase = blockIdx.x * 128;

  v8f acc[2][4];
  #pragma unroll
  for (int i = 0; i < 2; ++i)
    #pragma unroll
    for (int j = 0; j < 4; ++j)
      #pragma unroll
      for (int e = 0; e < 8; ++e) acc[i][j][e] = 0.0f;

  gemm_phase(acc, A1, lda1, M, W1, K1, N, rowBase, colBase, As, Bs);
  if (A2) gemm_phase(acc, A2, lda2, M, W2, K2, N, rowBase, colBase, As, Bs);

  const int tid  = threadIdx.x;
  const int lane = tid & 31;
  const int wave = tid >> 5;
  const int wm = wave >> 1, wn = wave & 1;
  const int lm = lane & 15;
  const int mo = (lane & 16) ? 8 : 0;   // C/D layout: lanes 16-31 hold M+8

  #pragma unroll
  for (int i = 0; i < 2; ++i)
    #pragma unroll
    for (int j = 0; j < 4; ++j) {
      const int n = colBase + wn * 64 + j * 16 + lm;
      if (n >= N) continue;
      float badd = 0.0f;
      if (b1) badd += b1[n];
      if (b2) badd += b2[n];
      #pragma unroll
      for (int v = 0; v < 8; ++v) {
        const int m = rowBase + wm * 32 + i * 16 + mo + v;
        if (m < M) {
          float x = acc[i][j][v] + badd;
          if (relu) x = fmaxf(x, 0.0f);
          C[(size_t)m * ldc + n] = x;
        }
      }
    }
}

// ---------------------------------------------------------------------------
// Elementwise / glue kernels
// ---------------------------------------------------------------------------
__global__ void convert_w_kernel(const float* __restrict__ W, unsigned short* __restrict__ Wp,
                                 int N, int K, int Kp)
{
  int idx = blockIdx.x * blockDim.x + threadIdx.x;
  if (idx >= N * Kp) return;
  int n = idx / Kp, k = idx - n * Kp;
  Wp[idx] = (k < K) ? f32_to_bf16_bits(W[(size_t)n * K + k]) : (unsigned short)0;
}

__global__ void zero_f32_kernel(float* __restrict__ p, size_t n) {
  size_t i = (size_t)blockIdx.x * blockDim.x + threadIdx.x;
  size_t stride = (size_t)gridDim.x * blockDim.x;
  for (; i < n; i += stride) p[i] = 0.0f;
}

// source (B,3,12,170) -> X0 (T,B,512 padded), cols 510..511 stay zero
__global__ void build_x0_kernel(const float* __restrict__ src, float* __restrict__ x0, int B) {
  int idx = blockIdx.x * blockDim.x + threadIdx.x;
  const int total = 12 * B * 510;
  if (idx >= total) return;
  int t = idx / (B * 510);
  int r = idx - t * (B * 510);
  int b = r / 510;
  int n = r - b * 510;
  int c = n / 170, f = n - c * 170;
  x0[((size_t)t * B + b) * 512 + n] = src[(((size_t)b * 3 + c) * 12 + t) * 170 + f];
}

// gates already include both biases (added in GEMM epilogue). PyTorch order i,f,g,o.
__global__ void lstm_cell_kernel(const float* __restrict__ gates, int ldg,
                                 float* __restrict__ c, float* __restrict__ h,
                                 int hid, float* __restrict__ yout, int ldy, int ycol, int B)
{
  int idx = blockIdx.x * blockDim.x + threadIdx.x;
  if (idx >= B * hid) return;
  int b = idx / hid, j = idx - b * hid;
  const float* g = gates + (size_t)b * ldg;
  float ig = g[j], fg = g[hid + j], gg = g[2 * hid + j], og = g[3 * hid + j];
  float si = 1.0f / (1.0f + __expf(-ig));
  float sf = 1.0f / (1.0f + __expf(-fg));
  float so = 1.0f / (1.0f + __expf(-og));
  float c2 = sf * c[idx] + si * tanhf(gg);
  float h2 = so * tanhf(c2);
  c[idx] = c2;
  h[idx] = h2;
  if (yout) yout[(size_t)b * ldy + ycol + j] = h2;
}

__global__ void copy_cols_kernel(float* __restrict__ dst, int ldd, int dcol,
                                 const float* __restrict__ src, int lds_, int scol,
                                 int rows, int cols)
{
  int idx = blockIdx.x * blockDim.x + threadIdx.x;
  if (idx >= rows * cols) return;
  int r = idx / cols, c0 = idx - r * cols;
  dst[(size_t)r * ldd + dcol + c0] = src[(size_t)r * lds_ + scol + c0];
}

// softmax over 12 attn logits (ldc 32), then applied[b,d] = sum_t aw*enc[t,b,d];
// writes directly into cat2 columns [510,1534).
__global__ void attn_apply_kernel(const float* __restrict__ logits,
                                  const float* __restrict__ enc,
                                  float* __restrict__ cat2, int B)
{
  int b = blockIdx.x;
  float aw[12];
  float mx = -1e30f;
  #pragma unroll
  for (int t = 0; t < 12; ++t) { aw[t] = logits[(size_t)b * 32 + t]; mx = fmaxf(mx, aw[t]); }
  float s = 0.0f;
  #pragma unroll
  for (int t = 0; t < 12; ++t) { aw[t] = __expf(aw[t] - mx); s += aw[t]; }
  float inv = 1.0f / s;
  #pragma unroll
  for (int t = 0; t < 12; ++t) aw[t] *= inv;
  for (int d = threadIdx.x; d < 1024; d += blockDim.x) {
    float a = 0.0f;
    #pragma unroll
    for (int t = 0; t < 12; ++t) a += aw[t] * enc[((size_t)t * B + b) * 1024 + d];
    cat2[(size_t)b * 1536 + 510 + d] = a;
  }
}

// pred (B,512 padded) -> out (B,3,12,170) at time t
__global__ void scatter_out_kernel(const float* __restrict__ pred, float* __restrict__ out,
                                   int t, int B)
{
  int idx = blockIdx.x * blockDim.x + threadIdx.x;
  if (idx >= B * 510) return;
  int b = idx / 510, n = idx - b * 510;
  int c = n / 170, f = n - c * 170;
  out[(((size_t)b * 3 + c) * 12 + t) * 170 + f] = pred[(size_t)b * 512 + n];
}

// ---------------------------------------------------------------------------
// Host orchestration (graph-capture safe: only launches on `stream`)
// Input flattening assumed depth-first dict order:
// 0:source 1:target_length 2..17:enc[l]{fwd,bwd}{Wih,Whh,bih,bhh}
// 18..25:dec[l]{Wih,Whh,bih,bhh} 26,27:attn{W,b} 28,29:attn_combine{W,b} 30,31:out{W,b}
// ---------------------------------------------------------------------------
extern "C" void kernel_launch(void* const* d_in, const int* in_sizes, int n_in,
                              void* d_out, int out_size, void* d_ws, size_t ws_size,
                              hipStream_t stream) {
  (void)in_sizes; (void)n_in; (void)out_size; (void)ws_size;
  const int B = 2048;
  const float* src = (const float*)d_in[0];
  int ii = 2;
  const float* eW[2][2][4];
  for (int l = 0; l < 2; ++l)
    for (int d = 0; d < 2; ++d)
      for (int p = 0; p < 4; ++p) eW[l][d][p] = (const float*)d_in[ii++];
  const float* dW[2][4];
  for (int l = 0; l < 2; ++l)
    for (int p = 0; p < 4; ++p) dW[l][p] = (const float*)d_in[ii++];
  const float* attnWf = (const float*)d_in[ii++];
  const float* attnBf = (const float*)d_in[ii++];
  const float* combWf = (const float*)d_in[ii++];
  const float* combBf = (const float*)d_in[ii++];
  const float* outWf  = (const float*)d_in[ii++];
  const float* outBf  = (const float*)d_in[ii++];

  size_t off = 0;
  auto alloc = [&](size_t bytes) -> void* {
    void* r = (char*)d_ws + off;
    off += (bytes + 255) & ~(size_t)255;
    return r;
  };

  // bf16 weights (K padded to mult of 32, zero-filled) — ~53 MB, L2-resident
  unsigned short *wihE[2][2], *whhE[2][2];
  for (int l = 0; l < 2; ++l)
    for (int d = 0; d < 2; ++d) {
      wihE[l][d] = (unsigned short*)alloc((size_t)2048 * (l == 0 ? 512 : 1024) * 2);
      whhE[l][d] = (unsigned short*)alloc((size_t)2048 * 512 * 2);
    }
  unsigned short *wihD[2], *whhD[2];
  wihD[0] = (unsigned short*)alloc((size_t)4096 * 512 * 2);
  whhD[0] = (unsigned short*)alloc((size_t)4096 * 1024 * 2);
  wihD[1] = (unsigned short*)alloc((size_t)4096 * 1024 * 2);
  whhD[1] = (unsigned short*)alloc((size_t)4096 * 1024 * 2);
  unsigned short* wAttn = (unsigned short*)alloc((size_t)12 * 1536 * 2);
  unsigned short* wComb = (unsigned short*)alloc((size_t)510 * 1536 * 2);
  unsigned short* wOut  = (unsigned short*)alloc((size_t)510 * 1024 * 2);

  // f32 activations (padded leading dims, pads zeroed each call)
  size_t actStart = off;
  float* X0 = (float*)alloc((size_t)12 * B * 512 * 4);
  float* Y0 = (float*)alloc((size_t)12 * B * 1024 * 4);
  float* Y1 = (float*)alloc((size_t)12 * B * 1024 * 4);   // enc_out
  float* gates = (float*)alloc((size_t)B * 4096 * 4);
  float *hE[4], *cE[4];
  for (int i = 0; i < 4; ++i) {
    hE[i] = (float*)alloc((size_t)B * 512 * 4);
    cE[i] = (float*)alloc((size_t)B * 512 * 4);
  }
  float *hD[2], *cD[2];
  for (int i = 0; i < 2; ++i) {
    hD[i] = (float*)alloc((size_t)B * 1024 * 4);
    cD[i] = (float*)alloc((size_t)B * 1024 * 4);
  }
  float* cat1   = (float*)alloc((size_t)B * 1536 * 4);
  float* cat2   = (float*)alloc((size_t)B * 1536 * 4);
  float* logits = (float*)alloc((size_t)B * 32 * 4);
  float* z      = (float*)alloc((size_t)B * 512 * 4);
  float* pred   = (float*)alloc((size_t)B * 512 * 4);
  size_t actFloats = (off - actStart) / 4;

  // deterministic: re-zero state/pads every call
  zero_f32_kernel<<<2048, 256, 0, stream>>>((float*)((char*)d_ws + actStart), actFloats);

  auto conv = [&](const float* Wsrc, unsigned short* Wdst, int N, int K) {
    int Kp = (K + 31) & ~31;
    int n = N * Kp;
    convert_w_kernel<<<(n + 255) / 256, 256, 0, stream>>>(Wsrc, Wdst, N, K, Kp);
  };
  for (int l = 0; l < 2; ++l)
    for (int d = 0; d < 2; ++d) {
      conv(eW[l][d][0], wihE[l][d], 2048, l == 0 ? 510 : 1024);
      conv(eW[l][d][1], whhE[l][d], 2048, 512);
    }
  conv(dW[0][0], wihD[0], 4096, 510);
  conv(dW[0][1], whhD[0], 4096, 1024);
  conv(dW[1][0], wihD[1], 4096, 1024);
  conv(dW[1][1], whhD[1], 4096, 1024);
  conv(attnWf, wAttn, 12, 1534);
  conv(combWf, wComb, 510, 1534);
  conv(outWf, wOut, 510, 1024);

  {
    int n = 12 * B * 510;
    build_x0_kernel<<<(n + 255) / 256, 256, 0, stream>>>(src, X0, B);
  }

  auto gemm = [&](float* C, int ldc, int M, int N,
                  const float* A1, int lda1, int K1, const unsigned short* W1,
                  const float* A2, int lda2, int K2, const unsigned short* W2,
                  const float* b1, const float* b2, int relu) {
    dim3 g((unsigned)((N + 127) / 128), (unsigned)((M + 127) / 128));
    gemm_wmma_kernel<<<g, 256, 0, stream>>>(C, ldc, M, N, A1, lda1, K1, W1,
                                            A2, lda2, K2, W2, b1, b2, relu);
  };
  auto cell = [&](const float* g, int ldg, float* c, float* h, int hid,
                  float* yout, int ldy, int ycol) {
    int n = B * hid;
    lstm_cell_kernel<<<(n + 255) / 256, 256, 0, stream>>>(g, ldg, c, h, hid, yout, ldy, ycol, B);
  };
  auto copyc = [&](float* dst, int ldd, int dcol, const float* s, int lds_, int scol,
                   int rows, int cols) {
    int n = rows * cols;
    copy_cols_kernel<<<(n + 255) / 256, 256, 0, stream>>>(dst, ldd, dcol, s, lds_, scol, rows, cols);
  };

  // ---------------- encoder: 2 layers x {fwd,bwd} ----------------
  for (int l = 0; l < 2; ++l) {
    const float* Xin = (l == 0) ? X0 : Y0;
    int ldx = (l == 0) ? 512 : 1024;
    float* Yout = (l == 0) ? Y0 : Y1;
    for (int d = 0; d < 2; ++d) {
      float* h = hE[2 * l + d];
      float* c = cE[2 * l + d];
      for (int s = 0; s < 12; ++s) {
        int t = (d == 0) ? s : (11 - s);
        gemm(gates, 2048, B, 2048,
             Xin + (size_t)t * B * ldx, ldx, ldx, wihE[l][d],
             h, 512, 512, whhE[l][d],
             eW[l][d][2], eW[l][d][3], 0);
        cell(gates, 2048, c, h, 512, Yout + (size_t)t * B * 1024, 1024, d * 512);
      }
    }
  }

  // decoder init: hall=[l0f,l0b,l1f,l1b]; h0[l]=cat(hall[l],hall[2+l])
  for (int l = 0; l < 2; ++l) {
    copyc(hD[l], 1024, 0,   hE[l],     512, 0, B, 512);
    copyc(hD[l], 1024, 512, hE[2 + l], 512, 0, B, 512);
    copyc(cD[l], 1024, 0,   cE[l],     512, 0, B, 512);
    copyc(cD[l], 1024, 512, cE[2 + l], 512, 0, B, 512);
  }

  // ---------------- decoder: 12 steps ----------------
  for (int t = 0; t < 12; ++t) {
    copyc(cat1, 1536, 0,   pred,  512,  0, B, 510);
    copyc(cat1, 1536, 510, hD[0], 1024, 0, B, 1024);
    gemm(logits, 32, B, 12, cat1, 1536, 1536, wAttn,
         nullptr, 0, 0, nullptr, attnBf, nullptr, 0);
    attn_apply_kernel<<<B, 256, 0, stream>>>(logits, Y1, cat2, B);
    copyc(cat2, 1536, 0, pred, 512, 0, B, 510);
    gemm(z, 512, B, 510, cat2, 1536, 1536, wComb,
         nullptr, 0, 0, nullptr, combBf, nullptr, 1);
    gemm(gates, 4096, B, 4096, z, 512, 512, wihD[0],
         hD[0], 1024, 1024, whhD[0], dW[0][2], dW[0][3], 0);
    cell(gates, 4096, cD[0], hD[0], 1024, nullptr, 0, 0);
    gemm(gates, 4096, B, 4096, hD[0], 1024, 1024, wihD[1],
         hD[1], 1024, 1024, whhD[1], dW[1][2], dW[1][3], 0);
    cell(gates, 4096, cD[1], hD[1], 1024, nullptr, 0, 0);
    gemm(pred, 512, B, 510, hD[1], 1024, 1024, wOut,
         nullptr, 0, 0, nullptr, outBf, nullptr, 0);
    int n = B * 510;
    scatter_out_kernel<<<(n + 255) / 256, 256, 0, stream>>>(pred, (float*)d_out, t, B);
  }
}
